// NRDE_classification_13048110645605
// MI455X (gfx1250) — matrix operand order repaired and checked by reading, not compiled, not run on hardware
//
#include <hip/hip_runtime.h>
#include <hip/hip_bf16.h>
#include <hip/hip_fp16.h>

// Problem constants (match reference)
#define B_    256
#define T_    2049
#define D_    64
#define H_    64
#define MLP_  128
#define NI_   128
#define OUT_  10
#define L_    2080          // D + D*(D-1)/2 ; == 65*32 (K divisible by 32)
#define S_    16            // (T-1)/NI
#define MD_   8192          // MLP_*D_
#define NPAIR 2016          // D*(D-1)/2
#define NC_   16            // RK4 steps per chunk (Wls chunk = [NC_*B_, MD_])
#define NCH_  8             // NI_/NC_
#define KSTEPS (L_ / 32)    // 65

typedef __attribute__((ext_vector_type(16))) _Float16 v16h;
typedef __attribute__((ext_vector_type(8)))  float    v8f;

#ifndef __has_builtin
#define __has_builtin(x) 0
#endif
#if __has_builtin(__builtin_amdgcn_global_load_async_to_lds_b128) && \
    __has_builtin(__builtin_amdgcn_s_wait_asynccnt)
#define USE_ASYNC_LDS 1
// Exact parameter types per hipcc diagnostic: pointer to GCC-vector int4,
// global (AS1) source and LDS (AS3) destination.
typedef int v4i_raw __attribute__((vector_size(4 * sizeof(int))));
typedef __attribute__((address_space(1))) v4i_raw* as1_v4i_p;
typedef __attribute__((address_space(3))) v4i_raw* as3_v4i_p;
#else
#define USE_ASYNC_LDS 0
#endif

// ---------------------------------------------------------------------------
// Elementwise fp32 -> f16 of W2 keeping its native [md][l] layout
// (flat index m*D*L + d*L + l == (m*64+d)*2080 + l). Per-column l-runs are
// contiguous, which is exactly what the GEMM's B-fragment (one column per
// lane, two runs of 8 K) wants — so no transpose is needed anywhere.
// ---------------------------------------------------------------------------
__global__ __launch_bounds__(256) void k_f2h(const float* __restrict__ src,
                                             _Float16* __restrict__ dst) {
  const size_t base = ((size_t)blockIdx.x * 256 + threadIdx.x) * 8;
  const float4 f0 = *(const float4*)(src + base);
  const float4 f1 = *(const float4*)(src + base + 4);
  union { _Float16 hx[8]; uint4 u; } o;
  o.hx[0] = (_Float16)f0.x; o.hx[1] = (_Float16)f0.y;
  o.hx[2] = (_Float16)f0.z; o.hx[3] = (_Float16)f0.w;
  o.hx[4] = (_Float16)f1.x; o.hx[5] = (_Float16)f1.y;
  o.hx[6] = (_Float16)f1.z; o.hx[7] = (_Float16)f1.w;
  *(uint4*)(dst + base) = o.u;
}

// ---------------------------------------------------------------------------
// Depth-2 log-signature. One WG per (b, n). n-major output:
// ls16[(n*B + b)*L + {0..63 = increments, 64+p = Levy pair p}] (triu order).
// ---------------------------------------------------------------------------
__global__ __launch_bounds__(256) void k_logsig(const float* __restrict__ X,
                                                _Float16* __restrict__ ls16) {
  const int bi = blockIdx.x;
  const int b  = bi >> 7;          // / NI
  const int n  = bi & (NI_ - 1);
  __shared__ float xs[(S_ + 1) * D_];   // 17*64
  const float* xb = X + ((size_t)b * T_ + (size_t)n * S_) * D_;
  for (int i = threadIdx.x; i < (S_ + 1) * D_; i += 256) xs[i] = xb[i];
  __syncthreads();

  const size_t row = ((size_t)n * B_ + b) * L_;
  if (threadIdx.x < D_) {
    const float inc = xs[S_ * D_ + threadIdx.x] - xs[threadIdx.x];
    ls16[row + threadIdx.x] = (_Float16)inc;
  }
  for (int p = threadIdx.x; p < NPAIR; p += 256) {
    // invert p = i*(127-i)/2 + (j-i-1)
    int i = (int)((127.0f - sqrtf(16129.0f - 8.0f * (float)p)) * 0.5f);
    if (i < 0) i = 0;
    while (i > 0 && (i * (127 - i)) / 2 > p) --i;
    while (((i + 1) * (126 - i)) / 2 <= p) ++i;
    const int j = i + 1 + (p - (i * (127 - i)) / 2);

    const float xi0 = xs[i], xj0 = xs[j];
    float acc = 0.f;
#pragma unroll
    for (int k = 0; k < S_; ++k) {
      const float si = xs[k * D_ + i];
      const float sj = xs[k * D_ + j];
      const float di = xs[(k + 1) * D_ + i] - si;
      const float dj = xs[(k + 1) * D_ + j] - sj;
      acc += (si - xi0) * dj - (sj - xj0) * di;
    }
    ls16[row + D_ + p] = (_Float16)(0.5f * acc);
  }
}

// ---------------------------------------------------------------------------
// bls[(n*B+b)*64 + d] = sum_l b2[d*L + l] * ls[n,b,l]  (step-invariant bias,
// hoisted out of the RK4 loop). One WG per (n,b); ls row staged in LDS.
// ---------------------------------------------------------------------------
__global__ __launch_bounds__(256) void k_bls(const _Float16* __restrict__ ls16,
                                             const float* __restrict__ b2,
                                             float* __restrict__ bls) {
  const int row = blockIdx.x;                 // 0..NI*B-1
  __shared__ float sls[L_];
  __shared__ float part[256];
  const _Float16* lsr = ls16 + (size_t)row * L_;
  for (int i = threadIdx.x; i < L_; i += 256) sls[i] = (float)lsr[i];
  __syncthreads();
  const int d = threadIdx.x >> 2;             // 0..63
  const int c = threadIdx.x & 3;              // 0..3, 520 l's each
  const float* br = b2 + (size_t)d * L_ + c * 520;
  float acc = 0.f;
  for (int i = 0; i < 520; ++i) acc += br[i] * sls[c * 520 + i];
  part[threadIdx.x] = acc;
  __syncthreads();
  if (c == 0)
    bls[(size_t)row * D_ + d] = part[threadIdx.x] + part[threadIdx.x + 1] +
                                part[threadIdx.x + 2] + part[threadIdx.x + 3];
}

// ---------------------------------------------------------------------------
// h0[b] = X[b,0,:] @ W_in + b_in
// ---------------------------------------------------------------------------
__global__ __launch_bounds__(64) void k_h0(const float* __restrict__ X,
                                           const float* __restrict__ W_in,
                                           const float* __restrict__ b_in,
                                           float* __restrict__ h) {
  const int b = blockIdx.x, j = threadIdx.x;
  const float* x0 = X + (size_t)b * T_ * D_;
  float acc = b_in[j];
  for (int i = 0; i < D_; ++i) acc += x0[i] * W_in[i * H_ + j];
  h[b * H_ + j] = acc;
}

// ---------------------------------------------------------------------------
// Chunked GEMM: Wls[NC*256, 8192]f32 = A(ls chunk)[NC*256, 2080]f16 @ B f16
// where B is W2h in native [md][l] layout (so B-tiles are contiguous 16B
// chunks too). Both tiles move via CDNA5 async global->LDS DMA into a
// double-buffered LDS (40 KB), overlapping DMA with WMMA; one barrier/K-step.
// 128x128 block tile, 8 waves -> each wave 32x64 = 2x4 WMMA tiles.
// ---------------------------------------------------------------------------
__global__ __launch_bounds__(256) void k_wls_gemm(const _Float16* __restrict__ A,
                                                  const _Float16* __restrict__ Bm,
                                                  float* __restrict__ C) {
  __shared__ _Float16 sA[2][128 * 40];   // [row][k], padded stride 40 f16
  __shared__ _Float16 sB[2][128 * 40];   // [col][k] (K-contiguous per column)

  const int tid  = threadIdx.x;
  const int lane = tid & 31;
  const int wave = tid >> 5;
  const int wm   = wave & 3;   // 32-row slab
  const int wn   = wave >> 2;  // 64-col slab
  const int mBase = blockIdx.y * 128;
  const int nBase = blockIdx.x * 128;

  v8f acc[2][4] = {};

  // 512 x 16B chunks per 8KB tile, 2 per thread: chunk c -> line c>>2, q=(c&3)*8
  const int c0 = tid * 2;

  auto issueTile = [&](int buf, int k0) {
#pragma unroll
    for (int e = 0; e < 2; ++e) {
      const int c    = c0 + e;
      const int line = c >> 2;
      const int q    = (c & 3) * 8;            // f16 offset within 64B line
#if USE_ASYNC_LDS
      __builtin_amdgcn_global_load_async_to_lds_b128(
          (as1_v4i_p)(A + (size_t)(mBase + line) * L_ + k0 + q),
          (as3_v4i_p)(&sA[buf][line * 40 + q]), 0, 0);
      __builtin_amdgcn_global_load_async_to_lds_b128(
          (as1_v4i_p)(Bm + (size_t)(nBase + line) * L_ + k0 + q),
          (as3_v4i_p)(&sB[buf][line * 40 + q]), 0, 0);
#else
      const uint4 va = *(const uint4*)(A + (size_t)(mBase + line) * L_ + k0 + q);
      *(uint4*)&sA[buf][line * 40 + q] = va;
      const uint4 vb = *(const uint4*)(Bm + (size_t)(nBase + line) * L_ + k0 + q);
      *(uint4*)&sB[buf][line * 40 + q] = vb;
#endif
    }
  };

  issueTile(0, 0);

  const int rl  = lane & 15;
  const int akb = (lane < 16) ? 0 : 8;   // A: lanes<16 K{0-7,16-23}; else K{8-15,24-31}
  const int bkb = (lane < 16) ? 0 : 8;   // B: lanes<16 column K{0-7,16-23}; else K{8-15,24-31}

  for (int kk = 0; kk < KSTEPS; ++kk) {
    const int cur = kk & 1;
#if USE_ASYNC_LDS
    __builtin_amdgcn_s_wait_asynccnt(0);
#endif
    __syncthreads();
    if (kk + 1 < KSTEPS) issueTile(cur ^ 1, (kk + 1) * 32);

    union Frag { uint4 u[2]; v16h h; };
    Frag af[2], bf[4];
#pragma unroll
    for (int t = 0; t < 2; ++t) {
      const int r = wm * 32 + t * 16 + rl;
      af[t].u[0] = *(const uint4*)&sA[cur][r * 40 + akb];
      af[t].u[1] = *(const uint4*)&sA[cur][r * 40 + akb + 16];
    }
#pragma unroll
    for (int u = 0; u < 4; ++u) {
      const int c = wn * 64 + u * 16 + rl;
      bf[u].u[0] = *(const uint4*)&sB[cur][c * 40 + bkb];
      bf[u].u[1] = *(const uint4*)&sB[cur][c * 40 + bkb + 16];
    }
#pragma unroll
    for (int t = 0; t < 2; ++t)
#pragma unroll
      for (int u = 0; u < 4; ++u)
        acc[t][u] = __builtin_amdgcn_wmma_f32_16x16x32_f16(
            false, af[t].h, false, bf[u].h, (short)0, acc[t][u], false, false);
  }

  // C/D layout: VGPR r -> row base + r + (lane<16 ? 0 : 8), col = base + (lane&15)
  const int rOff = (lane < 16) ? 0 : 8;
#pragma unroll
  for (int t = 0; t < 2; ++t) {
    const int rBase = mBase + wm * 32 + t * 16 + rOff;
#pragma unroll
    for (int u = 0; u < 4; ++u) {
      const int cc = nBase + wn * 64 + u * 16 + rl;
#pragma unroll
      for (int r = 0; r < 8; ++r)
        C[(size_t)(rBase + r) * MD_ + cc] = acc[t][u][r];
    }
  }
}

// ---------------------------------------------------------------------------
// RK4 over one chunk of NC_ steps. One WG (128 threads) per batch element;
// loops its steps internally (only true sequential dependence in the model).
// g(h) = tanh(h@W1 + b1) @ Wls[step,b] + bls[step,b]
// ---------------------------------------------------------------------------
__global__ __launch_bounds__(128) void k_rk4_chunk(const float* __restrict__ Wls,
                                                   const float* __restrict__ bls,
                                                   const float* __restrict__ W1,
                                                   const float* __restrict__ b1,
                                                   float* __restrict__ h) {
  const int b   = blockIdx.x;
  const int tid = threadIdx.x;
  __shared__ float sh[D_], shc[D_], sz[MLP_];

  if (tid < D_) {
    const float hv = h[b * D_ + tid];
    sh[tid]  = hv;
    shc[tid] = hv;
  }
  __syncthreads();

  const float wgt[4] = {1.f, 2.f, 2.f, 1.f};
  const float stp[4] = {0.5f, 0.5f, 1.f, 0.f};

  for (int s = 0; s < NC_; ++s) {
    const float* wbase = Wls + ((size_t)s * B_ + b) * MD_;
    float bl = 0.f;
    if (tid < D_) bl = bls[((size_t)s * B_ + b) * D_ + tid];

    float ksum = 0.f;
#pragma unroll
    for (int ev = 0; ev < 4; ++ev) {
      // z = tanh(hc @ W1 + b1)   (thread j owns output j)
      float a = b1[tid];
      for (int i = 0; i < D_; ++i) a += shc[i] * W1[i * MLP_ + tid];
      sz[tid] = tanhf(a);
      __syncthreads();
      if (tid < D_) {
        float g = bl;
        const float* wl = wbase + tid;
        for (int m = 0; m < MLP_; ++m) g += sz[m] * wl[m * D_];
        ksum += wgt[ev] * g;
        shc[tid] = sh[tid] + stp[ev] * g;
      }
      __syncthreads();
    }
    if (tid < D_) {
      const float hn = sh[tid] + ksum * (1.f / 6.f);
      sh[tid]  = hn;
      shc[tid] = hn;
    }
    __syncthreads();
  }
  if (tid < D_) h[b * D_ + tid] = sh[tid];
}

// ---------------------------------------------------------------------------
// out = hT @ W_out + b_out
// ---------------------------------------------------------------------------
__global__ __launch_bounds__(32) void k_out(const float* __restrict__ h,
                                            const float* __restrict__ W_out,
                                            const float* __restrict__ b_out,
                                            float* __restrict__ out) {
  const int b = blockIdx.x, o = threadIdx.x;
  if (o < OUT_) {
    float acc = b_out[o];
    for (int j = 0; j < H_; ++j) acc += h[b * H_ + j] * W_out[j * OUT_ + o];
    out[b * OUT_ + o] = acc;
  }
}

// ---------------------------------------------------------------------------
extern "C" void kernel_launch(void* const* d_in, const int* in_sizes, int n_in,
                              void* d_out, int out_size, void* d_ws, size_t ws_size,
                              hipStream_t stream) {
  const float* X     = (const float*)d_in[0];
  const float* W_in  = (const float*)d_in[1];
  const float* b_in  = (const float*)d_in[2];
  const float* W1    = (const float*)d_in[3];
  const float* b1    = (const float*)d_in[4];
  const float* W2    = (const float*)d_in[5];
  const float* b2    = (const float*)d_in[6];
  const float* W_out = (const float*)d_in[7];
  const float* b_out = (const float*)d_in[8];
  float* out = (float*)d_out;

  // workspace carve-out (~313 MB)
  char* ws = (char*)d_ws;
  size_t off = 0;
  _Float16* ls16 = (_Float16*)(ws + off); off += (size_t)NI_ * B_ * L_ * sizeof(_Float16);
  _Float16* W2h  = (_Float16*)(ws + off); off += (size_t)MD_ * L_ * sizeof(_Float16);
  float*    Wls  = (float*)(ws + off);    off += (size_t)NC_ * B_ * MD_ * sizeof(float);
  float*    bls  = (float*)(ws + off);    off += (size_t)NI_ * B_ * D_ * sizeof(float);
  float*    h    = (float*)(ws + off);    off += (size_t)B_ * H_ * sizeof(float);
  (void)off; (void)ws_size; (void)in_sizes; (void)n_in; (void)out_size;

  k_f2h<<<(MD_ * L_) / (256 * 8), 256, 0, stream>>>(W2, W2h);   // 8320 blocks, exact
  k_logsig<<<B_ * NI_, 256, 0, stream>>>(X, ls16);
  k_bls<<<NI_ * B_, 256, 0, stream>>>(ls16, b2, bls);
  k_h0<<<B_, 64, 0, stream>>>(X, W_in, b_in, h);

  for (int c = 0; c < NCH_; ++c) {
    const _Float16* lsc = ls16 + (size_t)c * NC_ * B_ * L_;
    k_wls_gemm<<<dim3(MD_ / 128, (NC_ * B_) / 128), 256, 0, stream>>>(lsc, W2h, Wls);
    k_rk4_chunk<<<B_, 128, 0, stream>>>(Wls, bls + (size_t)c * NC_ * B_ * D_, W1, b1, h);
  }

  k_out<<<B_, 32, 0, stream>>>(h, W_out, b_out, out);
}
